// GCN_38955353374769
// MI455X (gfx1250) — matrix operand order repaired and checked by reading, not compiled, not run on hardware
//
#include <hip/hip_runtime.h>

typedef __attribute__((ext_vector_type(16))) _Float16 v16h;
typedef __attribute__((ext_vector_type(8)))  _Float16 v8h;
typedef __attribute__((ext_vector_type(8)))  float    v8f;

#define DFEAT 512

// ---------------------------------------------------------------------------
// Fragment loaders (CDNA5 WMMA 16x16x32 f16 layouts, wave32)
// A (16x32, MxK): lane m = lane&15, kb = (lane>>4)*8; VGPR0-3 = K[kb..kb+7],
//                 VGPR4-7 = K[kb+16..kb+23]  (two contiguous 16B chunks)
// B (32x16, KxN): lane n = lane&15, kb = (lane>>4)*8; same K split as A
// C (16x16 f32):  lane n = lane&15; VGPR v -> M = v + (lane>=16 ? 8 : 0)
// ---------------------------------------------------------------------------

__device__ inline v16h load_a_frag(const _Float16* __restrict__ X, int lda,
                                   int m0, int k0, int lane) {
  const int m  = m0 + (lane & 15);
  const int kb = (lane >> 4) * 8;
  const _Float16* p = X + (size_t)m * lda + k0 + kb;
  v16h a;
#pragma unroll
  for (int t = 0; t < 8; ++t) a[t] = p[t];
#pragma unroll
  for (int t = 0; t < 8; ++t) a[8 + t] = p[16 + t];
  return a;
}

__device__ inline v16h load_b_frag(const _Float16* __restrict__ Bm, int ldb,
                                   int k0, int n0, int lane) {
  const int n  = n0 + (lane & 15);
  const int kb = (lane >> 4) * 8;
  const _Float16* p = Bm + (size_t)n * ldb + k0 + kb;
  v16h b;
#pragma unroll
  for (int t = 0; t < 8; ++t) b[t] = p[t];
#pragma unroll
  for (int t = 0; t < 8; ++t) b[8 + t] = p[16 + t];
  return b;
}

__device__ inline void zero_acc(v8f c[2][4]) {
#pragma unroll
  for (int i = 0; i < 2; ++i)
#pragma unroll
    for (int j = 0; j < 4; ++j)
#pragma unroll
      for (int t = 0; t < 8; ++t) c[i][j][t] = 0.0f;
}

// One 32(M) x 64(N) x 32(K) step: 2 A frags, 4 B frags, 8 WMMAs.
__device__ inline void mma_step(v8f c[2][4], const _Float16* __restrict__ A, int lda,
                                const _Float16* __restrict__ B, int ldb,
                                int m0, int n0, int k0, int lane) {
  v16h a0 = load_a_frag(A, lda, m0,      k0, lane);
  v16h a1 = load_a_frag(A, lda, m0 + 16, k0, lane);
#pragma unroll
  for (int j = 0; j < 4; ++j) {
    v16h b = load_b_frag(B, ldb, k0, n0 + 16 * j, lane);
    c[0][j] = __builtin_amdgcn_wmma_f32_16x16x32_f16(false, a0, false, b,
                                                     (short)0, c[0][j], false, false);
    c[1][j] = __builtin_amdgcn_wmma_f32_16x16x32_f16(false, a1, false, b,
                                                     (short)0, c[1][j], false, false);
  }
}

// ---------------------------------------------------------------------------
// GEMM1: infoT[b][f][j] = f16( gate[m] * sum_k X[m,k] * W[f,k] )
// X: f16 [Mtot,512] (global row m; batch = m / Dloc, j = m % Dloc)
// W: f16 [512,512] row-major (out, in) -> direct B fragments
// grid = (Mtot/32, 512/256), block = 128 (4 waves, 64 N-cols each)
// ---------------------------------------------------------------------------
__global__ void k_gemm1(const _Float16* __restrict__ X,
                        const _Float16* __restrict__ W,
                        const float* __restrict__ gate,
                        _Float16* __restrict__ outT, int Dloc) {
  const int lane = threadIdx.x & 31;
  const int wave = threadIdx.x >> 5;
  const int m0 = blockIdx.x * 32;
  const int n0 = blockIdx.y * 256 + wave * 64;

  v8f c[2][4];
  zero_acc(c);
  for (int k0 = 0; k0 < DFEAT; k0 += 32)
    mma_step(c, X, DFEAT, W, DFEAT, m0, n0, k0, lane);

  const int n  = lane & 15;
  const int hi = lane >> 4;
#pragma unroll
  for (int i = 0; i < 2; ++i) {
    const int mrow = m0 + 16 * i + hi * 8;   // 8-aligned; Dloc % 32 == 0
    const int bb   = mrow / Dloc;
    const int jloc = mrow % Dloc;
#pragma unroll
    for (int j = 0; j < 4; ++j) {
      const int f = n0 + 16 * j + n;
      v8h o;
#pragma unroll
      for (int v = 0; v < 8; ++v) o[v] = (_Float16)(c[i][j][v] * gate[mrow + v]);
      *(v8h*)(outT + ((size_t)bb * DFEAT + f) * Dloc + jloc) = o;
    }
  }
}

// ---------------------------------------------------------------------------
// GEMM2 (header aggregation): out[b][i][f] = sum_k A[b][i][k] * infoT[b][f][k]
// A: f16 0/1 relation, lda = K.  grid = (Mloc/32, 512/256, B), block = 128
// ---------------------------------------------------------------------------
__global__ void k_gemm2(const _Float16* __restrict__ A, const _Float16* __restrict__ BT,
                        float* __restrict__ out, int Mloc, int K) {
  const int lane = threadIdx.x & 31;
  const int wave = threadIdx.x >> 5;
  const int bz = blockIdx.z;
  const int m0 = blockIdx.x * 32;
  const int n0 = blockIdx.y * 256 + wave * 64;
  const _Float16* Ab = A + (size_t)bz * Mloc * K;
  const _Float16* Bb = BT + (size_t)bz * DFEAT * K;
  float* outb = out + (size_t)bz * Mloc * DFEAT;

  v8f c[2][4];
  zero_acc(c);
  for (int k0 = 0; k0 < K; k0 += 32) {
    if (k0 + 64 < K)
      __builtin_prefetch(Ab + (size_t)(m0 + (lane & 15)) * K + k0 + 64, 0, 1);
    mma_step(c, Ab, K, Bb, K, m0, n0, k0, lane);
  }

  const int n  = lane & 15;
  const int hi = lane >> 4;
#pragma unroll
  for (int i = 0; i < 2; ++i) {
    const int row = m0 + 16 * i + hi * 8;
#pragma unroll
    for (int j = 0; j < 4; ++j) {
      const int f = n0 + 16 * j + n;
      float* dst = outb + (size_t)row * DFEAT + f;
#pragma unroll
      for (int v = 0; v < 8; ++v) dst[(size_t)v * DFEAT] = c[i][j][v];
    }
  }
}

// ---------------------------------------------------------------------------
// Fused data-cell aggregation + combine:
//   c = samerow@infoSame + relTrow@infoRow + relTcol@infoCol   (x2 on the col
//       path is pre-folded into the gch gate)
//   v = relu( dm01 * c / dc_num );  write f16 state (+ f32 final output)
// grid = (2048/32, 512/256, B), block = 128
// ---------------------------------------------------------------------------
__global__ void k_aggdc(const _Float16* __restrict__ Asame, const _Float16* __restrict__ Bsame,
                        const _Float16* __restrict__ Arow,  const _Float16* __restrict__ Brow,
                        const _Float16* __restrict__ Acol,  const _Float16* __restrict__ Bcol,
                        const float* __restrict__ num, const int* __restrict__ mask,
                        _Float16* __restrict__ outH, float* __restrict__ outF) {
  constexpr int D = 2048, R = 64;
  const int lane = threadIdx.x & 31;
  const int wave = threadIdx.x >> 5;
  const int bz = blockIdx.z;
  const int m0 = blockIdx.x * 32;
  const int n0 = blockIdx.y * 256 + wave * 64;

  const _Float16* As = Asame + (size_t)bz * D * D;
  const _Float16* Bs = Bsame + (size_t)bz * DFEAT * D;
  const _Float16* Ar = Arow  + (size_t)bz * D * R;
  const _Float16* Br = Brow  + (size_t)bz * DFEAT * R;
  const _Float16* Ac = Acol  + (size_t)bz * D * R;
  const _Float16* Bc = Bcol  + (size_t)bz * DFEAT * R;

  v8f c[2][4];
  zero_acc(c);

  // same-row neighbors (dominant: K = 2048)
  for (int k0 = 0; k0 < D; k0 += 32) {
    if (k0 + 64 < D)
      __builtin_prefetch(As + (size_t)(m0 + (lane & 15)) * D + k0 + 64, 0, 1);
    mma_step(c, As, D, Bs, D, m0, n0, k0, lane);
  }
  // row headers (K = 64)
  for (int k0 = 0; k0 < R; k0 += 32)
    mma_step(c, Ar, R, Br, R, m0, n0, k0, lane);
  // col headers (K = 64; counted twice via doubled gate)
  for (int k0 = 0; k0 < R; k0 += 32)
    mma_step(c, Ac, R, Bc, R, m0, n0, k0, lane);

  const int n  = lane & 15;
  const int hi = lane >> 4;
#pragma unroll
  for (int i = 0; i < 2; ++i) {
    const int row = m0 + 16 * i + hi * 8;
#pragma unroll
    for (int j = 0; j < 4; ++j) {
      const int f = n0 + 16 * j + n;
#pragma unroll
      for (int v = 0; v < 8; ++v) {
        const size_t gm = (size_t)bz * D + row + v;
        float val = c[i][j][v];
        val = mask[gm] ? (val / num[gm]) : 0.0f;
        val = (val > 0.0f) ? val : 0.0f;
        const size_t idx = gm * DFEAT + f;
        outH[idx] = (_Float16)val;
        if (outF) outF[idx] = val;
      }
    }
  }
}

// ---------------------------------------------------------------------------
// Small helper kernels
// ---------------------------------------------------------------------------
__global__ void k_cvt_f16(const float* __restrict__ in, _Float16* __restrict__ out,
                          size_t n) {
  const size_t i = (size_t)blockIdx.x * blockDim.x + threadIdx.x;
  if (i < n) out[i] = (_Float16)in[i];
}

// int 0/1 relation -> exact f16 0/1 (one-time; relations are iteration-invariant)
__global__ void k_rel2h(const int* __restrict__ in, _Float16* __restrict__ out,
                        size_t n) {
  const size_t i = (size_t)blockIdx.x * blockDim.x + threadIdx.x;
  if (i < n) out[i] = (_Float16)(in[i] != 0 ? 1.0f : 0.0f);
}

// gate[m] = postScale * mask01[m] * sigmoid( dot(H[m,:], wn) ), one wave/node
__global__ void k_gate(const _Float16* __restrict__ H, const float* __restrict__ wn,
                       const int* __restrict__ mask, float* __restrict__ g, int M,
                       float postScale) {
  const int node = blockIdx.x * (blockDim.x >> 5) + (threadIdx.x >> 5);
  const int lane = threadIdx.x & 31;
  if (node >= M) return;
  const _Float16* row = H + (size_t)node * DFEAT;
  float acc = 0.0f;
#pragma unroll
  for (int t = 0; t < 16; ++t) {
    const int k = lane * 16 + t;
    acc += (float)row[k] * wn[k];
  }
#pragma unroll
  for (int off = 16; off >= 1; off >>= 1) acc += __shfl_xor(acc, off, 32);
  if (lane == 0) {
    const float s = 1.0f / (1.0f + expf(-acc));
    g[node] = mask[node] ? s * postScale : 0.0f;
  }
}

// out[row] = f(rowmask01 * sum_k (A[row,k] && colmask[k])), one wave per row
__global__ void k_rowsum(const int* __restrict__ A, const int* __restrict__ rowmask,
                         const int* __restrict__ colmask, float* __restrict__ out,
                         int Mloc, int K, int add, int clampMode) {
  const int row  = blockIdx.x * (blockDim.x >> 5) + (threadIdx.x >> 5);
  const int lane = threadIdx.x & 31;
  const int bb = row / Mloc;
  const int* Ar  = A + (size_t)row * K;
  const int* cmk = colmask + (size_t)bb * K;
  int acc = 0;
  for (int k = lane; k < K; k += 32) acc += (Ar[k] != 0 && cmk[k] != 0) ? 1 : 0;
#pragma unroll
  for (int off = 16; off >= 1; off >>= 1) acc += __shfl_xor(acc, off, 32);
  if (lane == 0) {
    const float s = rowmask[row] ? (float)acc : 0.0f;
    if (clampMode) out[row] = (s >= 1.0f) ? s + 1.0f : 1.0f;
    else if (add)  out[row] += s;
    else           out[row] = s;
  }
}

// out[b*D + j] += colmask01[j] * sum_r (A[b,r,j] && rowmask[r])
__global__ void k_colsum(const int* __restrict__ A, const int* __restrict__ rowmask,
                         const int* __restrict__ colmask, float* __restrict__ out,
                         int R, int D) {
  const int idx = blockIdx.x * blockDim.x + threadIdx.x;  // over B*D
  const int bb = idx / D;
  const int j  = idx % D;
  const int* Ab  = A + (size_t)bb * R * D;
  const int* rmk = rowmask + (size_t)bb * R;
  int acc = 0;
  for (int r = 0; r < R; ++r) acc += (Ab[(size_t)r * D + j] != 0 && rmk[r] != 0) ? 1 : 0;
  out[idx] += colmask[idx] ? (float)acc : 0.0f;
}

__global__ void k_clamp1(float* __restrict__ x, int n) {
  const int i = blockIdx.x * blockDim.x + threadIdx.x;
  if (i < n) { const float v = x[i]; x[i] = (v >= 1.0f) ? v : 1.0f; }
}

// qinfo[b,f] = sum_k q[b,k] * W[f,k]   (tiny, fp32, iteration-invariant)
__global__ void k_qinfo(const float* __restrict__ q, const float* __restrict__ W,
                        float* __restrict__ out) {
  const int idx = blockIdx.x * blockDim.x + threadIdx.x;  // B*512
  const int bb = idx >> 9;
  const int f  = idx & 511;
  const float* qb = q + (size_t)bb * DFEAT;
  const float* wr = W + (size_t)f * DFEAT;
  float acc = 0.0f;
  for (int k = 0; k < DFEAT; ++k) acc += qb[k] * wr[k];
  out[idx] = acc;
}

// relT[b][i][r] = f16( rel[b][r][i] != 0 )
__global__ void k_trel(const int* __restrict__ rel, _Float16* __restrict__ outT,
                       int R, int D) {
  const size_t idx = (size_t)blockIdx.x * blockDim.x + threadIdx.x;  // B*D*R
  const int r = (int)(idx % R);
  const size_t t = idx / R;
  const int i = (int)(t % D);
  const int bb = (int)(t / D);
  outT[idx] = (_Float16)((rel[((size_t)bb * R + r) * D + i] != 0) ? 1.0f : 0.0f);
}

// v = relu( mask01[m] * (agg + qinfo[b,f]) / num[m] ); write f16 state (+f32 out)
__global__ void k_combine(const float* __restrict__ agg, const float* __restrict__ qinf,
                          const float* __restrict__ num, const int* __restrict__ mask,
                          _Float16* __restrict__ outH, float* __restrict__ outF,
                          int Mloc) {
  const size_t idx = (size_t)blockIdx.x * blockDim.x + threadIdx.x;
  const int m = (int)(idx >> 9);
  const int f = (int)(idx & 511);
  float v = agg[idx];
  const int bb = m / Mloc;
  v += qinf[(size_t)bb * DFEAT + f];
  v = mask[m] ? (v / num[m]) : 0.0f;
  v = (v > 0.0f) ? v : 0.0f;
  outH[idx] = (_Float16)v;
  if (outF) outF[idx] = v;
}

// ---------------------------------------------------------------------------
extern "C" void kernel_launch(void* const* d_in, const int* in_sizes, int n_in,
                              void* d_out, int out_size, void* d_ws, size_t ws_size,
                              hipStream_t stream) {
  (void)in_sizes; (void)n_in; (void)out_size; (void)ws_size;
  constexpr int Bn = 16, R = 64, C = 64, D = 2048;
  constexpr int MRH = Bn * R;        // 1024
  constexpr int MDC = Bn * D;        // 32768

  const float* q       = (const float*)d_in[0];
  const float* rh0     = (const float*)d_in[1];
  const float* ch0     = (const float*)d_in[2];
  const float* dc0     = (const float*)d_in[3];
  const int*   rm      = (const int*)d_in[4];
  const int*   cm      = (const int*)d_in[5];
  const int*   dm      = (const int*)d_in[6];
  const int*   samerow = (const int*)d_in[7];
  const int*   samecol = (const int*)d_in[8];
  const int*   rowrel  = (const int*)d_in[9];    // [B,R,D]
  const int*   colrel  = (const int*)d_in[10];   // [B,C,D]
  const float* w_node  = (const float*)d_in[11];
  const float* wRow    = (const float*)d_in[12];
  const float* wCol    = (const float*)d_in[13];
  const float* wDcRow  = (const float*)d_in[14];
  const float* wDcCol  = (const float*)d_in[15];
  const float* wSameR  = (const float*)d_in[16];
  // d_in[17] (w_same_col) is dead code in the reference output path
  const float* wRowQ   = (const float*)d_in[18];
  const float* wColQ   = (const float*)d_in[19];

  char* ws = (char*)d_ws;
  size_t off = 0;
  auto walloc = [&](size_t bytes) -> char* {
    char* p = ws + off;
    off += (bytes + 255) & ~(size_t)255;
    return p;
  };
  _Float16* wRow16   = (_Float16*)walloc((size_t)DFEAT * DFEAT * 2);
  _Float16* wCol16   = (_Float16*)walloc((size_t)DFEAT * DFEAT * 2);
  _Float16* wSame16  = (_Float16*)walloc((size_t)DFEAT * DFEAT * 2);
  _Float16* wDcRow16 = (_Float16*)walloc((size_t)DFEAT * DFEAT * 2);
  _Float16* wDcCol16 = (_Float16*)walloc((size_t)DFEAT * DFEAT * 2);
  _Float16* dcH      = (_Float16*)walloc((size_t)MDC * DFEAT * 2);
  _Float16* rhH      = (_Float16*)walloc((size_t)MRH * DFEAT * 2);
  _Float16* chH      = (_Float16*)walloc((size_t)MRH * DFEAT * 2);
  float*    gdc      = (float*)walloc((size_t)MDC * 4);
  float*    grh      = (float*)walloc((size_t)MRH * 4);
  float*    gch      = (float*)walloc((size_t)MRH * 4);
  float*    rh_num   = (float*)walloc((size_t)MRH * 4);
  float*    ch_num   = (float*)walloc((size_t)MRH * 4);
  float*    dc_num   = (float*)walloc((size_t)MDC * 4);
  float*    qrow     = (float*)walloc((size_t)Bn * DFEAT * 4);
  float*    qcol     = (float*)walloc((size_t)Bn * DFEAT * 4);
  _Float16* samerowH = (_Float16*)walloc((size_t)Bn * D * D * 2);   // 134 MB
  _Float16* rowrelH  = (_Float16*)walloc((size_t)Bn * R * D * 2);
  _Float16* colrelH  = (_Float16*)walloc((size_t)Bn * C * D * 2);
  _Float16* relTrow  = (_Float16*)walloc((size_t)Bn * D * R * 2);
  _Float16* relTcol  = (_Float16*)walloc((size_t)Bn * D * C * 2);
  _Float16* infoTbig = (_Float16*)walloc((size_t)Bn * DFEAT * D * 2);
  _Float16* infoTr   = (_Float16*)walloc((size_t)Bn * DFEAT * R * 2);
  _Float16* infoTc   = (_Float16*)walloc((size_t)Bn * DFEAT * C * 2);
  float*    agg_rh   = (float*)walloc((size_t)MRH * DFEAT * 4);
  float*    agg_ch   = (float*)walloc((size_t)MRH * DFEAT * 4);

  float* outF = (float*)d_out;
  float* out_rh = outF;
  float* out_ch = outF + (size_t)MRH * DFEAT;
  float* out_dc = outF + (size_t)2 * MRH * DFEAT;

  // ---- one-time prep -----------------------------------------------------
  const size_t nW = (size_t)DFEAT * DFEAT;
  k_cvt_f16<<<(unsigned)((nW + 255) / 256), 256, 0, stream>>>(wRow,   wRow16,   nW);
  k_cvt_f16<<<(unsigned)((nW + 255) / 256), 256, 0, stream>>>(wCol,   wCol16,   nW);
  k_cvt_f16<<<(unsigned)((nW + 255) / 256), 256, 0, stream>>>(wSameR, wSame16,  nW);
  k_cvt_f16<<<(unsigned)((nW + 255) / 256), 256, 0, stream>>>(wDcRow, wDcRow16, nW);
  k_cvt_f16<<<(unsigned)((nW + 255) / 256), 256, 0, stream>>>(wDcCol, wDcCol16, nW);

  const size_t nDC = (size_t)MDC * DFEAT, nRH = (size_t)MRH * DFEAT;
  k_cvt_f16<<<(unsigned)((nDC + 255) / 256), 256, 0, stream>>>(dc0, dcH, nDC);
  k_cvt_f16<<<(unsigned)((nRH + 255) / 256), 256, 0, stream>>>(rh0, rhH, nRH);
  k_cvt_f16<<<(unsigned)((nRH + 255) / 256), 256, 0, stream>>>(ch0, chH, nRH);

  // exact f16 copies of the 0/1 relation graphs (A-operands for WMMA)
  const size_t nSR = (size_t)Bn * D * D;
  const size_t nRR = (size_t)Bn * R * D;
  k_rel2h<<<(unsigned)(nSR / 256), 256, 0, stream>>>(samerow, samerowH, nSR);
  k_rel2h<<<(unsigned)(nRR / 256), 256, 0, stream>>>(rowrel,  rowrelH,  nRR);
  k_rel2h<<<(unsigned)(nRR / 256), 256, 0, stream>>>(colrel,  colrelH,  nRR);

  k_qinfo<<<(Bn * DFEAT) / 256, 256, 0, stream>>>(q, wRowQ, qrow);
  k_qinfo<<<(Bn * DFEAT) / 256, 256, 0, stream>>>(q, wColQ, qcol);

  // degree counts (iteration-invariant)
  k_rowsum<<<MRH / 8, 256, 0, stream>>>(rowrel, rm, dm, rh_num, R, D, 0, 1);
  k_rowsum<<<MRH / 8, 256, 0, stream>>>(colrel, cm, dm, ch_num, C, D, 0, 1);
  k_rowsum<<<MDC / 8, 256, 0, stream>>>(samerow, dm, dm, dc_num, D, D, 0, 0);
  k_rowsum<<<MDC / 8, 256, 0, stream>>>(samecol, dm, dm, dc_num, D, D, 1, 0);
  k_colsum<<<MDC / 256, 256, 0, stream>>>(rowrel, rm, dm, dc_num, R, D);
  k_colsum<<<MDC / 256, 256, 0, stream>>>(colrel, cm, dm, dc_num, C, D);
  k_clamp1<<<MDC / 256, 256, 0, stream>>>(dc_num, MDC);

  // transposed 0/1 relations for dc <- header aggregation
  const size_t nT = (size_t)Bn * D * R;
  k_trel<<<(unsigned)(nT / 256), 256, 0, stream>>>(rowrel, relTrow, R, D);
  k_trel<<<(unsigned)(nT / 256), 256, 0, stream>>>(colrel, relTcol, C, D);

  // ---- 2 message-passing iterations --------------------------------------
  for (int it = 0; it < 2; ++it) {
    const bool last = (it == 1);

    k_gate<<<MDC / 8, 256, 0, stream>>>(dcH, w_node, dm, gdc, MDC, 1.0f);
    k_gate<<<MRH / 8, 256, 0, stream>>>(rhH, w_node, rm, grh, MRH, 1.0f);
    // NOTE: reference sums dc_ch_agg TWICE -> fold x2 into the gch gate
    k_gate<<<MRH / 8, 256, 0, stream>>>(chH, w_node, cm, gch, MRH, 2.0f);

    dim3 blk(128);
    dim3 gDC(MDC / 32, DFEAT / 256);        // dense transforms on data cells
    dim3 gRH(MRH / 32, DFEAT / 256);        // dense transforms on headers
    dim3 gA_rh(R / 32, DFEAT / 256, Bn);    // agg into headers
    dim3 gA_dc(D / 32, DFEAT / 256, Bn);    // fused agg into data cells

    // header <- data-cell aggregation (row)
    k_gemm1<<<gDC, blk, 0, stream>>>(dcH, wRow16, gdc, infoTbig, D);
    k_gemm2<<<gA_rh, blk, 0, stream>>>(rowrelH, infoTbig, agg_rh, R, D);
    // header <- data-cell aggregation (col)
    k_gemm1<<<gDC, blk, 0, stream>>>(dcH, wCol16, gdc, infoTbig, D);
    k_gemm2<<<gA_rh, blk, 0, stream>>>(colrelH, infoTbig, agg_ch, C, D);
    // data-cell update: all three neighbor transforms, then one fused kernel
    k_gemm1<<<gDC, blk, 0, stream>>>(dcH, wSame16, gdc, infoTbig, D);
    k_gemm1<<<gRH, blk, 0, stream>>>(rhH, wDcRow16, grh, infoTr, R);
    k_gemm1<<<gRH, blk, 0, stream>>>(chH, wDcCol16, gch, infoTc, C);
    k_aggdc<<<gA_dc, blk, 0, stream>>>(samerowH, infoTbig, relTrow, infoTr,
                                       relTcol, infoTc, dc_num, dm, dcH,
                                       last ? out_dc : nullptr);

    k_combine<<<(unsigned)(nRH / 256), 256, 0, stream>>>(
        agg_rh, qrow, rh_num, rm, rhH, last ? out_rh : nullptr, R);
    k_combine<<<(unsigned)(nRH / 256), 256, 0, stream>>>(
        agg_ch, qcol, ch_num, cm, chH, last ? out_ch : nullptr, C);
  }
}
